// KairosModel_69793218560130
// MI455X (gfx1250) — matrix operand order repaired and checked by reading, not compiled, not run on hardware
//
#include <hip/hip_runtime.h>
#include <math.h>

typedef __bf16 bf16;
typedef __attribute__((ext_vector_type(16))) __bf16 v16bf;
typedef __attribute__((ext_vector_type(8)))  __bf16 v8bf;
typedef __attribute__((ext_vector_type(4)))  __bf16 v4bf;
typedef __attribute__((ext_vector_type(8)))  float  v8f;
typedef __attribute__((ext_vector_type(4)))  int    v4i32;

#define N_NODES 50000
#define N_EDGES 150000
#define D_FEAT  100
#define E_FEAT  200
#define H1_FEAT 800

#define BM 128
#define BN 64
#define BK 32

#ifndef __has_builtin
#define __has_builtin(x) 0
#endif
#if __has_builtin(__builtin_amdgcn_global_load_async_to_lds_b128)
#define HAVE_ASYNC_LDS 1
#else
#define HAVE_ASYNC_LDS 0
#endif

static __device__ __forceinline__ float bf2f(bf16 b) {
  unsigned short s = __builtin_bit_cast(unsigned short, b);
  return __builtin_bit_cast(float, (unsigned)s << 16);
}
static __device__ __forceinline__ bf16 f2bf(float f) {
  unsigned u = __builtin_bit_cast(unsigned, f);
  u += 0x7fffu + ((u >> 16) & 1u);          // round to nearest even
  return __builtin_bit_cast(bf16, (unsigned short)(u >> 16));
}
static __device__ __forceinline__ void atomic_max_f32(float* addr, float v) {
  if (v >= 0.0f) atomicMax((int*)addr, __float_as_int(v));
  else           atomicMin((unsigned int*)addr, __float_as_uint(v));
}
static __device__ __forceinline__ v8bf zero_v8bf() {
  v8bf z;
#pragma unroll
  for (int j = 0; j < 8; ++j) z[j] = f2bf(0.0f);
  return z;
}

#if HAVE_ASYNC_LDS
typedef __attribute__((address_space(1))) v4i32 as1_v4i;
typedef __attribute__((address_space(3))) v4i32 as3_v4i;
static __device__ __forceinline__ void async_b128(const bf16* g, bf16* l) {
  __builtin_amdgcn_global_load_async_to_lds_b128((as1_v4i*)g, (as3_v4i*)l, 0, 0);
}
#endif

// ---------------------------------------------------------------- elementwise
__global__ void fill_f32_kernel(float* __restrict__ p, float v, long long n) {
  long long i = (long long)blockIdx.x * blockDim.x + threadIdx.x;
  if (i < n) p[i] = v;
}

__global__ void relu_bf16_kernel(const float* __restrict__ in, bf16* __restrict__ out,
                                 long long n) {
  long long i = (long long)blockIdx.x * blockDim.x + threadIdx.x;
  if (i < n) out[i] = f2bf(fmaxf(in[i], 0.0f));
}

// f32 [R,C] -> bf16 [Rpad, ldo], zero pad (used for x and all weights)
__global__ void cvt_pad_bf16_kernel(const float* __restrict__ in, bf16* __restrict__ out,
                                    int R, int C, int ldo, long long total) {
  long long i = (long long)blockIdx.x * blockDim.x + threadIdx.x;
  if (i >= total) return;
  int row = (int)(i / ldo);
  int col = (int)(i % ldo);
  float v = (row < R && col < C) ? in[(long long)row * C + col] : 0.0f;
  out[i] = f2bf(v);
}

// edge_attr[e, 0:100] = cos(rel_t*time_w+time_b) ; [100:200] = msg ; [200:ld) = 0
__global__ void edge_attr_kernel(const int* __restrict__ last_update,
                                 const int* __restrict__ tt,
                                 const int* __restrict__ src,
                                 const float* __restrict__ time_w,
                                 const float* __restrict__ time_b,
                                 const float* __restrict__ msg,
                                 bf16* __restrict__ out, int ld) {
  long long i = (long long)blockIdx.x * blockDim.x + threadIdx.x;
  const long long total = (long long)N_EDGES * ld;
  if (i >= total) return;
  int e = (int)(i / ld);
  int j = (int)(i % ld);
  float v = 0.0f;
  if (j < D_FEAT) {
    float rel = (float)(last_update[src[e]] - tt[e]);
    v = cosf(rel * time_w[j] + time_b[j]);
  } else if (j < E_FEAT) {
    v = msg[(long long)e * D_FEAT + (j - D_FEAT)];
  }
  out[i] = f2bf(v);
}

// ---------------------------------------------------------------- WMMA GEMM
// C = act(A[row(m)] @ B + bias). All bf16 buffers padded to 32-col multiples,
// zero-filled pad -> no per-element K guards. Kpad = align32(K) <= lda, and B
// has align32(K) zero-padded rows. Writes zeros for n in [N, nstore).
// Double-buffered LDS; A tile staged via async global->LDS when available.
__global__ __launch_bounds__(256)
void gemm_bf16_wmma_kernel(const bf16* __restrict__ A, const int* __restrict__ Aidx,
                           int lda, int Kpad,
                           const bf16* __restrict__ B, int ldb,
                           const float* __restrict__ bias,
                           bf16* __restrict__ Cbf, float* __restrict__ Cf,
                           int M, int N, int ldc, int col_off, int nstore, int act) {
  __shared__ __attribute__((aligned(16))) bf16 As[2][BM][BK + 8];   // row-major
  __shared__ __attribute__((aligned(16))) bf16 Bs[2][BN][BK + 8];   // transposed [n][k]

  const int tid  = threadIdx.x;
  const int lane = tid & 31;
  const int wave = tid >> 5;       // 0..7
  const int wm   = wave >> 1;      // 0..3 -> 32-row slab
  const int wn   = wave & 1;       // 0..1 -> 32-col slab

  const int blockM = blockIdx.y * BM;
  const int blockN = blockIdx.x * BN;

  v8f acc00 = {}, acc01 = {}, acc10 = {}, acc11 = {};

  // A loader: row = tid>>1 (0..127), 16 halves at (tid&1)*16
  const int arow_l = tid >> 1;
  const int acol0  = (tid & 1) * 16;
  long long asrc = -1;
  {
    int arow_g = blockM + arow_l;
    if (arow_g < M) asrc = Aidx ? (long long)Aidx[arow_g] : (long long)arow_g;
  }
  // B loader: k = tid&31, 8 halves at n0 = (tid>>5)*8 (transpose-store into Bs)
  const int bk  = tid & 31;
  const int bn0 = (tid >> 5) * 8;
  const int nG  = blockN + bn0;
  const bool bin = (nG + 8 <= ldb);

  const int nsteps = Kpad / BK;

  auto load_b_regs = [&](int k0) -> v8bf {
    if (bin) return *(const v8bf*)(B + (long long)(k0 + bk) * ldb + nG);
    return zero_v8bf();
  };
  auto stage_a = [&](int k0, int buf) {
    if (asrc >= 0) {
      const bf16* gp = A + asrc * (long long)lda + k0 + acol0;
#if HAVE_ASYNC_LDS
      async_b128(gp,     &As[buf][arow_l][acol0]);
      async_b128(gp + 8, &As[buf][arow_l][acol0 + 8]);
#else
      const v8bf* g = (const v8bf*)gp;
      v8bf a0 = g[0], a1 = g[1];
      if (k0 + BK < Kpad) __builtin_prefetch(gp + BK, 0, 3);
      *(v8bf*)&As[buf][arow_l][acol0]     = a0;
      *(v8bf*)&As[buf][arow_l][acol0 + 8] = a1;
#endif
    } else {
      v8bf z = zero_v8bf();
      *(v8bf*)&As[buf][arow_l][acol0]     = z;
      *(v8bf*)&As[buf][arow_l][acol0 + 8] = z;
    }
  };
  auto store_b = [&](v8bf bv, int buf) {
#pragma unroll
    for (int j = 0; j < 8; ++j) Bs[buf][bn0 + j][bk] = bv[j];
  };

  // ---- prologue: stage tile 0 into buffer 0 ----
  stage_a(0, 0);
  store_b(load_b_regs(0), 0);
#if HAVE_ASYNC_LDS
  asm volatile("s_wait_asynccnt 0x0" ::: "memory");
#endif
  __syncthreads();

  for (int s = 0; s < nsteps; ++s) {
    const int cur  = s & 1;
    const int nxt  = cur ^ 1;
    const bool more = (s + 1) < nsteps;

    // issue next tile's global traffic early (overlaps the WMMAs below)
    v8bf nb = zero_v8bf();
    if (more) {
      stage_a((s + 1) * BK, nxt);     // async path writes LDS directly
      nb = load_b_regs((s + 1) * BK);
    }

    // ---- fragments: each = two contiguous 16B chunks -> ds_load_b128 ----
    const int mrow = wm * 32 + (lane & 15);
    const int kba  = (lane >> 4) * 8;    // A: halves j<8 at kba, j>=8 at kba+16
    const int ncol = wn * 32 + (lane & 15);
    const int kbb  = (lane >> 4) * 8;    // B: halves j<8 at kbb, j>=8 at kbb+16
    v16bf af0, af1, bf0, bf1;
    {
      v8bf lo = *(const v8bf*)&As[cur][mrow][kba];
      v8bf hi = *(const v8bf*)&As[cur][mrow][kba + 16];
#pragma unroll
      for (int j = 0; j < 8; ++j) { af0[j] = lo[j]; af0[j + 8] = hi[j]; }
      lo = *(const v8bf*)&As[cur][mrow + 16][kba];
      hi = *(const v8bf*)&As[cur][mrow + 16][kba + 16];
#pragma unroll
      for (int j = 0; j < 8; ++j) { af1[j] = lo[j]; af1[j + 8] = hi[j]; }
      lo = *(const v8bf*)&Bs[cur][ncol][kbb];
      hi = *(const v8bf*)&Bs[cur][ncol][kbb + 16];
#pragma unroll
      for (int j = 0; j < 8; ++j) { bf0[j] = lo[j]; bf0[j + 8] = hi[j]; }
      lo = *(const v8bf*)&Bs[cur][ncol + 16][kbb];
      hi = *(const v8bf*)&Bs[cur][ncol + 16][kbb + 16];
#pragma unroll
      for (int j = 0; j < 8; ++j) { bf1[j] = lo[j]; bf1[j + 8] = hi[j]; }
    }
    acc00 = __builtin_amdgcn_wmma_f32_16x16x32_bf16(false, af0, false, bf0, (short)0, acc00, false, false);
    acc01 = __builtin_amdgcn_wmma_f32_16x16x32_bf16(false, af0, false, bf1, (short)0, acc01, false, false);
    acc10 = __builtin_amdgcn_wmma_f32_16x16x32_bf16(false, af1, false, bf0, (short)0, acc10, false, false);
    acc11 = __builtin_amdgcn_wmma_f32_16x16x32_bf16(false, af1, false, bf1, (short)0, acc11, false, false);

    if (more) {
      store_b(nb, nxt);
#if HAVE_ASYNC_LDS
      asm volatile("s_wait_asynccnt 0x0" ::: "memory");
#endif
      __syncthreads();
    }
  }

  // ---- store: lanes 0-15: M=vgpr,N=lane ; lanes 16-31: M=8+vgpr,N=lane-16 ----
  const int n_l  = lane & 15;
  const int m_hi = (lane >> 4) * 8;
#pragma unroll
  for (int a = 0; a < 2; ++a) {
#pragma unroll
    for (int i = 0; i < 8; ++i) {
      int m = blockM + wm * 32 + a * 16 + m_hi + i;
      if (m >= M) continue;
#pragma unroll
      for (int b = 0; b < 2; ++b) {
        int n = blockN + wn * 32 + b * 16 + n_l;
        if (n < nstore) {
          float v = 0.0f;
          if (n < N) {
            v = (a == 0) ? ((b == 0) ? acc00[i] : acc01[i])
                         : ((b == 0) ? acc10[i] : acc11[i]);
            if (bias) v += bias[n];
            if (act == 1)      v = fmaxf(v, 0.0f);
            else if (act == 2) v = tanhf(v);
          }
          long long o = (long long)m * ldc + col_off + n;
          if (Cf) Cf[o] = v;
          else    Cbf[o] = f2bf(v);
        }
      }
    }
  }
}

// ---------------------------------------------------------------- attention
__global__ void attn_alpha_kernel(const bf16* __restrict__ q, const bf16* __restrict__ k,
                                  const bf16* __restrict__ ep,
                                  const int* __restrict__ src, const int* __restrict__ dst,
                                  float* __restrict__ alpha, float* __restrict__ amax,
                                  int H, int C, int ld, float scale) {
  long long gid = (long long)blockIdx.x * blockDim.x + threadIdx.x;
  if (gid >= (long long)N_EDGES * H) return;
  int e = (int)(gid / H), h = (int)(gid % H);
  int s = src[e], d = dst[e];
  const v4bf* qr = (const v4bf*)(q  + (long long)d * ld + h * C);
  const v4bf* kr = (const v4bf*)(k  + (long long)s * ld + h * C);
  const v4bf* er = (const v4bf*)(ep + (long long)e * ld + h * C);
  float acc = 0.0f;
  for (int c4 = 0; c4 < C / 4; ++c4) {
    v4bf qa = qr[c4], ka = kr[c4], ea = er[c4];
#pragma unroll
    for (int j = 0; j < 4; ++j) acc += bf2f(qa[j]) * (bf2f(ka[j]) + bf2f(ea[j]));
  }
  acc *= scale;
  alpha[gid] = acc;
  atomic_max_f32(&amax[(long long)d * H + h], acc);
}

__global__ void attn_exp_kernel(const int* __restrict__ dst, float* __restrict__ alpha,
                                const float* __restrict__ amax, float* __restrict__ denom,
                                int H) {
  long long gid = (long long)blockIdx.x * blockDim.x + threadIdx.x;
  if (gid >= (long long)N_EDGES * H) return;
  int e = (int)(gid / H), h = (int)(gid % H);
  float ex = __expf(alpha[gid] - amax[(long long)dst[e] * H + h]);
  alpha[gid] = ex;
  atomicAdd(&denom[(long long)dst[e] * H + h], ex);
}

// one block per edge; s_acc already holds the skip connection (x@sw + sb)
__global__ void attn_agg_kernel(const bf16* __restrict__ v, const bf16* __restrict__ ep,
                                const int* __restrict__ src, const int* __restrict__ dst,
                                const float* __restrict__ ex, const float* __restrict__ denom,
                                float* __restrict__ s_acc, int H, int C, int ld) {
  int e = blockIdx.x;
  int s = src[e], d = dst[e];
  const int HC = H * C;
  const v4bf* vr = (const v4bf*)(v  + (long long)s * ld);
  const v4bf* er = (const v4bf*)(ep + (long long)e * ld);
  float* sr = s_acc + (long long)d * ld;
  for (int j4 = threadIdx.x; j4 * 4 < HC; j4 += blockDim.x) {
    int j = j4 * 4;
    int h = j / C;
    float a = ex[(long long)e * H + h] / (denom[(long long)d * H + h] + 1e-16f);
    v4bf va = vr[j4], ea = er[j4];
#pragma unroll
    for (int t = 0; t < 4; ++t) {
      // C==100 is a multiple of 4, so all 4 elements share head h
      atomicAdd(&sr[j + t], a * (bf2f(va[t]) + bf2f(ea[t])));
    }
  }
}

// ---------------------------------------------------------------- host side
static inline int ald32(int x) { return (x + 31) & ~31; }
static inline int cdivll(long long a, int b) { return (int)((a + b - 1) / b); }

static void launch_gemm(const bf16* A, const int* Aidx, int K, int lda,
                        const bf16* B, int ldb, const float* bias,
                        bf16* Cbf, float* Cf, int M, int N, int ldc, int col_off,
                        int nstore, int act, hipStream_t stream) {
  dim3 grid((N + BN - 1) / BN, (M + BM - 1) / BM);
  gemm_bf16_wmma_kernel<<<grid, 256, 0, stream>>>(A, Aidx, lda, ald32(K), B, ldb, bias,
                                                  Cbf, Cf, M, N, ldc, col_off, nstore, act);
}

static void launch_cvt_pad(const float* in, bf16* out, int R, int C, int ldo, int Rpad,
                           hipStream_t stream) {
  long long total = (long long)Rpad * ldo;
  cvt_pad_bf16_kernel<<<cdivll(total, 256), 256, 0, stream>>>(in, out, R, C, ldo, total);
}

static void run_tconv(hipStream_t stream,
                      const bf16* xin, int Kin, int H, int C,
                      const bf16* wq, const float* bq, const bf16* wk, const float* bk,
                      const bf16* wv, const float* bv, const bf16* wsk, const float* bsk,
                      const bf16* we, const bf16* eattr, int ld_ea,
                      const int* src, const int* dst,
                      bf16* qb, bf16* kb_, bf16* vb, bf16* eb,
                      float* sb, float* alpha, float* amax, float* denom,
                      bf16* hout) {
  const int HC  = H * C;
  const int lin = ald32(Kin);
  const int ldo = ald32(HC);
  launch_gemm(xin,   nullptr, Kin,    lin,   wq,  ldo, bq,  qb,  nullptr, N_NODES, HC, ldo, 0, ldo, 0, stream);
  launch_gemm(xin,   nullptr, Kin,    lin,   wk,  ldo, bk,  kb_, nullptr, N_NODES, HC, ldo, 0, ldo, 0, stream);
  launch_gemm(xin,   nullptr, Kin,    lin,   wv,  ldo, bv,  vb,  nullptr, N_NODES, HC, ldo, 0, ldo, 0, stream);
  launch_gemm(xin,   nullptr, Kin,    lin,   wsk, ldo, bsk, nullptr, sb,  N_NODES, HC, ldo, 0, ldo, 0, stream);
  launch_gemm(eattr, nullptr, E_FEAT, ld_ea, we,  ldo, nullptr, eb, nullptr, N_EDGES, HC, ldo, 0, ldo, 0, stream);

  long long nh = (long long)N_NODES * H;
  fill_f32_kernel<<<cdivll(nh, 256), 256, 0, stream>>>(amax, -INFINITY, nh);
  fill_f32_kernel<<<cdivll(nh, 256), 256, 0, stream>>>(denom, 0.0f, nh);

  long long eh = (long long)N_EDGES * H;
  float scale = 1.0f / sqrtf((float)C);
  attn_alpha_kernel<<<cdivll(eh, 256), 256, 0, stream>>>(qb, kb_, eb, src, dst,
                                                         alpha, amax, H, C, ldo, scale);
  attn_exp_kernel<<<cdivll(eh, 256), 256, 0, stream>>>(dst, alpha, amax, denom, H);
  attn_agg_kernel<<<N_EDGES, 256, 0, stream>>>(vb, eb, src, dst, alpha, denom, sb, H, C, ldo);

  long long nn = (long long)N_NODES * ldo;
  relu_bf16_kernel<<<cdivll(nn, 256), 256, 0, stream>>>(sb, hout, nn);
}

extern "C" void kernel_launch(void* const* d_in, const int* in_sizes, int n_in,
                              void* d_out, int out_size, void* d_ws, size_t ws_size,
                              hipStream_t stream) {
  (void)in_sizes; (void)n_in; (void)out_size; (void)ws_size;
  const float* x      = (const float*)d_in[0];
  const float* msg    = (const float*)d_in[1];
  const float* time_w = (const float*)d_in[2];
  const float* time_b = (const float*)d_in[3];
  const float* c1_qw = (const float*)d_in[4];  const float* c1_qb = (const float*)d_in[5];
  const float* c1_kw = (const float*)d_in[6];  const float* c1_kb = (const float*)d_in[7];
  const float* c1_vw = (const float*)d_in[8];  const float* c1_vb = (const float*)d_in[9];
  const float* c1_ew = (const float*)d_in[10];
  const float* c1_sw = (const float*)d_in[11]; const float* c1_sb = (const float*)d_in[12];
  const float* c2_qw = (const float*)d_in[13]; const float* c2_qb = (const float*)d_in[14];
  const float* c2_kw = (const float*)d_in[15]; const float* c2_kb = (const float*)d_in[16];
  const float* c2_vw = (const float*)d_in[17]; const float* c2_vb = (const float*)d_in[18];
  const float* c2_ew = (const float*)d_in[19];
  const float* c2_sw = (const float*)d_in[20]; const float* c2_sb = (const float*)d_in[21];
  const float* lps_w = (const float*)d_in[22]; const float* lps_b = (const float*)d_in[23];
  const float* lpd_w = (const float*)d_in[24]; const float* lpd_b = (const float*)d_in[25];
  const float* lp1_w = (const float*)d_in[26]; const float* lp1_b = (const float*)d_in[27];
  const float* lp2_w = (const float*)d_in[28]; const float* lp2_b = (const float*)d_in[29];
  const float* lp3_w = (const float*)d_in[30]; const float* lp3_b = (const float*)d_in[31];
  const float* lp4_w = (const float*)d_in[32]; const float* lp4_b = (const float*)d_in[33];
  const int* last_update = (const int*)d_in[34];
  const int* tt          = (const int*)d_in[35];
  const int* ei          = (const int*)d_in[36];
  const int* src = ei;
  const int* dst = ei + N_EDGES;

  // padded leading dims
  const int L100 = ald32(D_FEAT);    // 128
  const int L200 = ald32(E_FEAT);    // 224
  const int L800 = ald32(H1_FEAT);   // 800
  const int L400 = ald32(4 * D_FEAT);// 416
  const int L50  = ald32(D_FEAT / 2);// 64
  const int L2   = ald32(2);         // 32

  // ---- workspace bump allocator -------------------------------------------
  char* base = (char*)d_ws;
  size_t off = 0;
  auto alloc = [&](size_t bytes) -> void* {
    void* p = base + off;
    off += (bytes + 255) & ~(size_t)255;
    return p;
  };

  // persistent: bf16 x, padded bf16 weights, edge_attr, h1, h2
  bf16* xbf   = (bf16*)alloc((size_t)N_NODES * L100 * 2);
  bf16* w_c1q = (bf16*)alloc((size_t)L100 * L800 * 2);
  bf16* w_c1k = (bf16*)alloc((size_t)L100 * L800 * 2);
  bf16* w_c1v = (bf16*)alloc((size_t)L100 * L800 * 2);
  bf16* w_c1s = (bf16*)alloc((size_t)L100 * L800 * 2);
  bf16* w_c1e = (bf16*)alloc((size_t)L200 * L800 * 2);
  bf16* w_c2q = (bf16*)alloc((size_t)L800 * L100 * 2);
  bf16* w_c2k = (bf16*)alloc((size_t)L800 * L100 * 2);
  bf16* w_c2v = (bf16*)alloc((size_t)L800 * L100 * 2);
  bf16* w_c2s = (bf16*)alloc((size_t)L800 * L100 * 2);
  bf16* w_c2e = (bf16*)alloc((size_t)L200 * L100 * 2);
  bf16* w_lps = (bf16*)alloc((size_t)L100 * L200 * 2);
  bf16* w_lpd = (bf16*)alloc((size_t)L100 * L200 * 2);
  bf16* w_lp1 = (bf16*)alloc((size_t)L400 * L800 * 2);
  bf16* w_lp2 = (bf16*)alloc((size_t)L800 * L200 * 2);
  bf16* w_lp3 = (bf16*)alloc((size_t)L200 * L50 * 2);
  bf16* w_lp4 = (bf16*)alloc((size_t)L50 * L2 * 2);
  bf16* eattr = (bf16*)alloc((size_t)N_EDGES * L200 * 2);
  bf16* h1    = (bf16*)alloc((size_t)N_NODES * L800 * 2);
  bf16* h2    = (bf16*)alloc((size_t)N_NODES * L100 * 2);
  const size_t scratch0 = off;

  // ---- conversions (zero-padded) ------------------------------------------
  launch_cvt_pad(x,     xbf,   N_NODES, D_FEAT, L100, N_NODES, stream);
  launch_cvt_pad(c1_qw, w_c1q, D_FEAT, H1_FEAT, L800, L100, stream);
  launch_cvt_pad(c1_kw, w_c1k, D_FEAT, H1_FEAT, L800, L100, stream);
  launch_cvt_pad(c1_vw, w_c1v, D_FEAT, H1_FEAT, L800, L100, stream);
  launch_cvt_pad(c1_sw, w_c1s, D_FEAT, H1_FEAT, L800, L100, stream);
  launch_cvt_pad(c1_ew, w_c1e, E_FEAT, H1_FEAT, L800, L200, stream);
  launch_cvt_pad(c2_qw, w_c2q, H1_FEAT, D_FEAT, L100, L800, stream);
  launch_cvt_pad(c2_kw, w_c2k, H1_FEAT, D_FEAT, L100, L800, stream);
  launch_cvt_pad(c2_vw, w_c2v, H1_FEAT, D_FEAT, L100, L800, stream);
  launch_cvt_pad(c2_sw, w_c2s, H1_FEAT, D_FEAT, L100, L800, stream);
  launch_cvt_pad(c2_ew, w_c2e, E_FEAT, D_FEAT, L100, L200, stream);
  launch_cvt_pad(lps_w, w_lps, D_FEAT, 2 * D_FEAT, L200, L100, stream);
  launch_cvt_pad(lpd_w, w_lpd, D_FEAT, 2 * D_FEAT, L200, L100, stream);
  launch_cvt_pad(lp1_w, w_lp1, 4 * D_FEAT, 8 * D_FEAT, L800, L400, stream);
  launch_cvt_pad(lp2_w, w_lp2, 8 * D_FEAT, 2 * D_FEAT, L200, L800, stream);
  launch_cvt_pad(lp3_w, w_lp3, 2 * D_FEAT, D_FEAT / 2, L50, L200, stream);
  launch_cvt_pad(lp4_w, w_lp4, D_FEAT / 2, 2, L2, L50, stream);

  // ---- edge attributes -----------------------------------------------------
  edge_attr_kernel<<<cdivll((long long)N_EDGES * L200, 256), 256, 0, stream>>>(
      last_update, tt, src, time_w, time_b, msg, eattr, L200);

  // ---- TransformerConv 1 (H=8, C=100, in=100) -----------------------------
  {
    off = scratch0;
    bf16*  q1    = (bf16*)alloc((size_t)N_NODES * L800 * 2);
    bf16*  k1    = (bf16*)alloc((size_t)N_NODES * L800 * 2);
    bf16*  v1    = (bf16*)alloc((size_t)N_NODES * L800 * 2);
    bf16*  e1    = (bf16*)alloc((size_t)N_EDGES * L800 * 2);
    float* s1    = (float*)alloc((size_t)N_NODES * L800 * 4);
    float* alpha = (float*)alloc((size_t)N_EDGES * 8 * 4);
    float* amax  = (float*)alloc((size_t)N_NODES * 8 * 4);
    float* denom = (float*)alloc((size_t)N_NODES * 8 * 4);
    run_tconv(stream, xbf, D_FEAT, 8, D_FEAT,
              w_c1q, c1_qb, w_c1k, c1_kb, w_c1v, c1_vb, w_c1s, c1_sb, w_c1e,
              eattr, L200, src, dst, q1, k1, v1, e1, s1, alpha, amax, denom, h1);
  }

  // ---- TransformerConv 2 (H=1, C=100, in=800, concat=False) ---------------
  {
    off = scratch0;
    bf16*  q2    = (bf16*)alloc((size_t)N_NODES * L100 * 2);
    bf16*  k2    = (bf16*)alloc((size_t)N_NODES * L100 * 2);
    bf16*  v2    = (bf16*)alloc((size_t)N_NODES * L100 * 2);
    bf16*  e2    = (bf16*)alloc((size_t)N_EDGES * L100 * 2);
    float* s2    = (float*)alloc((size_t)N_NODES * L100 * 4);
    float* alpha = (float*)alloc((size_t)N_EDGES * 4);
    float* amax  = (float*)alloc((size_t)N_NODES * 4);
    float* denom = (float*)alloc((size_t)N_NODES * 4);
    run_tconv(stream, h1, H1_FEAT, 1, D_FEAT,
              w_c2q, c2_qb, w_c2k, c2_kb, w_c2v, c2_vb, w_c2s, c2_sb, w_c2e,
              eattr, L200, src, dst, q2, k2, v2, e2, s2, alpha, amax, denom, h2);
  }

  // ---- Link predictor ------------------------------------------------------
  {
    off = scratch0;
    bf16* hh = (bf16*)alloc((size_t)N_EDGES * L400 * 2);  // [E,400] (ld 416)
    bf16* t1 = (bf16*)alloc((size_t)N_EDGES * L800 * 2);  // [E,800]
    bf16* t2 = (bf16*)alloc((size_t)N_EDGES * L200 * 2);  // [E,200] (ld 224)
    bf16* t3 = (bf16*)alloc((size_t)N_EDGES * L50 * 2);   // [E,50]  (ld 64)

    // hh = [h2[src]@Wsrc + b | h2[dst]@Wdst + b]; dst part launched second so
    // its writes win over the src part's zero padding in cols [200,224)
    launch_gemm(h2, src, D_FEAT, L100, w_lps, L200, lps_b, hh, nullptr,
                N_EDGES, 2 * D_FEAT, L400, 0, L200, 0, stream);
    launch_gemm(h2, dst, D_FEAT, L100, w_lpd, L200, lpd_b, hh, nullptr,
                N_EDGES, 2 * D_FEAT, L400, 2 * D_FEAT, L400 - 2 * D_FEAT, 0, stream);
    // tanh MLP chain
    launch_gemm(hh, nullptr, 4 * D_FEAT, L400, w_lp1, L800, lp1_b, t1, nullptr,
                N_EDGES, 8 * D_FEAT, L800, 0, L800, 2, stream);
    launch_gemm(t1, nullptr, 8 * D_FEAT, L800, w_lp2, L200, lp2_b, t2, nullptr,
                N_EDGES, 2 * D_FEAT, L200, 0, L200, 2, stream);
    launch_gemm(t2, nullptr, 2 * D_FEAT, L200, w_lp3, L50, lp3_b, t3, nullptr,
                N_EDGES, D_FEAT / 2, L50, 0, L50, 2, stream);
    // final logits, f32 -> d_out (nstore = N = 2: never writes outside output)
    launch_gemm(t3, nullptr, D_FEAT / 2, L50, w_lp4, L2, lp4_b, nullptr, (float*)d_out,
                N_EDGES, 2, 2, 0, 2, 0, stream);
  }
}
